// FourierResidualLayer32_18124761989257
// MI455X (gfx1250) — matrix operand order repaired and checked by reading, not compiled, not run on hardware
//
#include <hip/hip_runtime.h>

typedef __attribute__((ext_vector_type(16))) _Float16 v16h;
typedef __attribute__((ext_vector_type(8)))  _Float16 v8h;
typedef __attribute__((ext_vector_type(4)))  _Float16 v4h;
typedef __attribute__((ext_vector_type(8)))  float    v8f;
typedef __attribute__((ext_vector_type(4)))  float    v4f;

#define DIM     512
#define M_TILE  128
#define LDK     (DIM + 8)      // padded feat row stride (halves) -> bank de-conflict
#define INV_2PI 0.15915494309189535f

// ---- kernel 1: w2 [K][N] f32  ->  w2T [N][K] f16 (workspace) ----------------
__global__ __launch_bounds__(256) void w2_transpose_f16(
    const float* __restrict__ w2, _Float16* __restrict__ w2t) {
  int idx = blockIdx.x * 256 + threadIdx.x;   // 512*512 elements
  int k = idx >> 9;
  int n = idx & (DIM - 1);
  w2t[(size_t)n * DIM + k] = (_Float16)w2[idx];
}

// ---- kernel 2: feat = cos(x*w1+b1) -> LDS f16;  out = feat @ w2 + x ---------
__global__ __launch_bounds__(256) void fourier_residual_kernel(
    const float* __restrict__ x, const float* __restrict__ w1,
    const float* __restrict__ b1, const _Float16* __restrict__ w2t,
    float* __restrict__ out) {
  __shared__ _Float16 sFeat[M_TILE * LDK];   // ~130 KB
  __shared__ float    sW1[DIM];
  __shared__ float    sB1[DIM];

  const int tid  = threadIdx.x;
  const int row0 = blockIdx.x * M_TILE;

  // stage w1/b1 (2 floats per thread)
  {
    int c = tid * 2;
    sW1[c]     = w1[c];
    sW1[c + 1] = w1[c + 1];
    sB1[c]     = b1[c];
    sB1[c + 1] = b1[c + 1];
  }
  __syncthreads();

  // Phase 1: 128x512 features, float4 chunks (64 per thread), coalesced x reads.
  // cos(a) == v_cos(v_fract(a * 1/2pi))  (hardware cos takes revolutions)
#pragma unroll 4
  for (int i = 0; i < 64; ++i) {
    int c  = i * 256 + tid;
    int r  = c >> 7;            // 128 float4-chunks per row
    int cc = (c & 127) << 2;    // column
    v4f xv = *(const v4f*)(x + (size_t)(row0 + r) * DIM + cc);
    v4f wv = *(const v4f*)(sW1 + cc);
    v4f bv = *(const v4f*)(sB1 + cc);
    v4h h;
#pragma unroll
    for (int j = 0; j < 4; ++j) {
      float rev = __builtin_amdgcn_fractf((xv[j] * wv[j] + bv[j]) * INV_2PI);
      h[j] = (_Float16)__builtin_amdgcn_cosf(rev);
    }
    *(v4h*)(sFeat + r * LDK + cc) = h;
  }
  __syncthreads();

  const int wave    = tid >> 5;     // 8 waves (wave32)
  const int lane    = tid & 31;
  const int half_id = lane >> 4;    // K-half selector for A/B fragments
  const int ln      = lane & 15;    // M row (A) / N col (B,C,D)

  // Two LDS base pointers so every A-fragment ds_load uses a static immediate
  // offset < 64KB (mt 0..3 off apLo, mt 4..7 off apHi). This removes the
  // in-loop v_add address math and with it the WMMA->VALU WAR hazard v_nops.
  const _Float16* apLo = sFeat + ln * LDK + half_id * 8;
  unsigned hiOff = 64u * LDK;          // 64 rows ahead (halves)
  asm volatile("" : "+v"(hiOff));      // opaque: keep two distinct bases
  const _Float16* apHi = apLo + hiOff;

  // Each wave owns 64 output columns: 2 passes of a 32-column pair.
  // One A fragment feeds two WMMAs (both column tiles) -> half the LDS traffic.
#pragma clang loop unroll(disable)
  for (int cp = 0; cp < 2; ++cp) {
    const int col0 = wave * 64 + cp * 32;
    v8f acc0[8] = {};   // columns [col0,      col0+16)
    v8f acc1[8] = {};   // columns [col0+16,   col0+32)

#pragma clang loop unroll(disable)
    for (int kt = 0; kt < 16; ++kt) {
      const int kbase = kt * 32;
      // B fragments (32x16): lane(half,n) elems 0..7 = K kbase+half*8+0..7, elems 8..15 = K kbase+16+half*8+0..7
      const _Float16* bp0 = w2t + (size_t)(col0 + ln) * DIM + kbase + half_id * 8;
      const _Float16* bp1 = bp0 + (size_t)16 * DIM;
      v8h b0l = *(const v8h*)(bp0);
      v8h b0h = *(const v8h*)(bp0 + 16);
      v8h b1l = *(const v8h*)(bp1);
      v8h b1h = *(const v8h*)(bp1 + 16);
      v16h bf0 = __builtin_shufflevector(b0l, b0h, 0, 1, 2, 3, 4, 5, 6, 7, 8, 9,
                                         10, 11, 12, 13, 14, 15);
      v16h bf1 = __builtin_shufflevector(b1l, b1h, 0, 1, 2, 3, 4, 5, 6, 7, 8, 9,
                                         10, 11, 12, 13, 14, 15);
      // A fragment (16x32): lane(half,m=ln); elems 0..7 = K kbase+half*8+0..7,
      // elems 8..15 = K kbase+16+half*8+0..7
#pragma unroll
      for (int mt = 0; mt < 4; ++mt) {
        const _Float16* ap = apLo + kbase + mt * 16 * LDK;
        v8h a0 = *(const v8h*)(ap);
        v8h a1 = *(const v8h*)(ap + 16);
        v16h af = __builtin_shufflevector(a0, a1, 0, 1, 2, 3, 4, 5, 6, 7, 8, 9,
                                          10, 11, 12, 13, 14, 15);
        acc0[mt] = __builtin_amdgcn_wmma_f32_16x16x32_f16(
            false, af, false, bf0, (short)0, acc0[mt], false, false);
        acc1[mt] = __builtin_amdgcn_wmma_f32_16x16x32_f16(
            false, af, false, bf1, (short)0, acc1[mt], false, false);
      }
#pragma unroll
      for (int mt = 0; mt < 4; ++mt) {
        const _Float16* ap = apHi + kbase + mt * 16 * LDK;
        v8h a0 = *(const v8h*)(ap);
        v8h a1 = *(const v8h*)(ap + 16);
        v16h af = __builtin_shufflevector(a0, a1, 0, 1, 2, 3, 4, 5, 6, 7, 8, 9,
                                          10, 11, 12, 13, 14, 15);
        acc0[mt + 4] = __builtin_amdgcn_wmma_f32_16x16x32_f16(
            false, af, false, bf0, (short)0, acc0[mt + 4], false, false);
        acc1[mt + 4] = __builtin_amdgcn_wmma_f32_16x16x32_f16(
            false, af, false, bf1, (short)0, acc1[mt + 4], false, false);
      }
    }

    // Epilogue: D element r -> (row0 + mt*16 + half*8 + r, col), fused +x.
    // Batch all 16 x-loads of an M-tile (immediate offsets off one base) so
    // they stay in flight together, then add+store. Avoids 1-load-at-a-time
    // serialization; x/out are __restrict__ so batches are independent.
    const size_t ebase = (size_t)(row0 + half_id * 8) * DIM + col0 + ln;
    const float* xb = x + ebase;
    float*       ob = out + ebase;
#pragma unroll
    for (int mt = 0; mt < 8; ++mt) {
      float xv0[8], xv1[8];
#pragma unroll
      for (int r = 0; r < 8; ++r) {
        int off = (mt * 16 + r) * DIM;
        xv0[r] = xb[off];
        xv1[r] = xb[off + 16];
      }
#pragma unroll
      for (int r = 0; r < 8; ++r) {
        int off = (mt * 16 + r) * DIM;
        ob[off]      = acc0[mt][r] + xv0[r];
        ob[off + 16] = acc1[mt][r] + xv1[r];
      }
    }
  }
}

extern "C" void kernel_launch(void* const* d_in, const int* in_sizes, int n_in,
                              void* d_out, int out_size, void* d_ws, size_t ws_size,
                              hipStream_t stream) {
  const float* x  = (const float*)d_in[0];
  const float* w1 = (const float*)d_in[1];
  const float* w2 = (const float*)d_in[2];
  const float* b1 = (const float*)d_in[3];
  float* out      = (float*)d_out;
  _Float16* w2t   = (_Float16*)d_ws;   // 512*512*2 = 512 KB

  const int nrows = in_sizes[0] / DIM;          // 131072

  w2_transpose_f16<<<(DIM * DIM) / 256, 256, 0, stream>>>(w2, w2t);
  fourier_residual_kernel<<<nrows / M_TILE, 256, 0, stream>>>(x, w1, b1, w2t, out);
}
